// GCNModel_1546188226613
// MI455X (gfx1250) — matrix operand order repaired and checked by reading, not compiled, not run on hardware
//
#include <hip/hip_runtime.h>
#include <hip/hip_bf16.h>
#include <stdint.h>

// ---------------- problem constants (match reference) ----------------
#define NNODES 50000
#define NEDGES 600000
#define NGRAPH 64
#define CH     256      // hidden width of both GCN layers

typedef __attribute__((ext_vector_type(16))) __bf16 v16bf;
typedef __attribute__((ext_vector_type(8)))  float  v8f;

union ABFrag { v16bf v; uint32_t u[8]; uint4 q[2]; };

__device__ __forceinline__ uint32_t f2bf(float f) {
  uint32_t u = __float_as_uint(f);
  u += 0x7FFFu + ((u >> 16) & 1u);          // round-to-nearest-even
  return u >> 16;
}

// ---------------- conversion / packing kernels ----------------
// f32 row-major -> bf16 pairs packed in dwords (same row-major order)
__global__ void k_f32_to_bf16(const float* __restrict__ X, uint32_t* __restrict__ Xb,
                              int totalPairs) {
  int i = blockIdx.x * blockDim.x + threadIdx.x;
  if (i >= totalPairs) return;
  Xb[i] = f2bf(X[2 * i]) | (f2bf(X[2 * i + 1]) << 16);
}

// Pack W[K][256] f32 into fragment-contiguous bf16 layout:
//   Wp dword index = ((kStep*16 + nTile)*32 + lane)*8 + v
// so a wave's B fragment for (kStep, nTile) is 8 consecutive dwords per lane
// (two b128 loads). Element mapping uses contiguous k order (k = 16*half + element), same as the A loads:
//   half = lane>>4, lh = lane&15, pair p = kStep*16 + half*8 + v  (K = 2p, 2p+1)
//   n = nTile*16 + lh
__global__ void k_pack_w_frag(const float* __restrict__ W, uint32_t* __restrict__ Wp,
                              int K, int Nw) {
  int i = blockIdx.x * blockDim.x + threadIdx.x;
  if (i >= (K / 2) * Nw) return;
  int v    = i & 7;
  int lane = (i >> 3) & 31;
  int tn   = (i >> 8) & 15;
  int kk   = i >> 12;
  int half = lane >> 4, lh = lane & 15;
  int p = kk * 16 + half * 8 + v;
  int n = tn * 16 + lh;
  Wp[i] = f2bf(W[(2 * p) * Nw + n]) | (f2bf(W[(2 * p + 1) * Nw + n]) << 16);
}

// ---------------- degree / normalization ----------------
__global__ void k_deg_init(float* __restrict__ deg, int n) {
  int i = blockIdx.x * blockDim.x + threadIdx.x;
  if (i < n) deg[i] = 1.0f;                  // self loop
}
__global__ void k_deg_edges(const long long* __restrict__ ei, float* __restrict__ deg, int E) {
  int e = blockIdx.x * blockDim.x + threadIdx.x;
  if (e >= E) return;
  atomicAdd(&deg[(int)ei[E + e]], 1.0f);     // dst row of edge_index
}
__global__ void k_dinv(const float* __restrict__ deg, float* __restrict__ dinv, int n) {
  int i = blockIdx.x * blockDim.x + threadIdx.x;
  if (i < n) dinv[i] = __frsqrt_rn(deg[i]);
}

// ---------------- WMMA GEMM: C[M][256] = A[M][K](bf16) * Wp(fragment-packed bf16) ----------
// One wave computes a 16(M) x 64(N) tile; 4 v_wmma per K-step sharing one A fragment.
// Per K-step VMEM: 2 x b128 (A) + 8 x b128 (B) thanks to fragment-contiguous weights.
template <int K>
__global__ void __launch_bounds__(256) k_wmma_gemm(const uint16_t* __restrict__ A,
                                                   const uint32_t* __restrict__ Bp,
                                                   float* __restrict__ C, int M) {
  const int NW = CH;                          // 256
  int wid  = (blockIdx.x * blockDim.x + threadIdx.x) >> 5;
  int lane = threadIdx.x & 31;
  int half = lane >> 4, lh = lane & 15;
  int mTiles = M >> 4;
  if (wid >= mTiles * (NW / 64)) return;      // uniform per wave: EXEC stays all-ones
  int mt      = wid / (NW / 64);
  int ntBase  = (wid % (NW / 64)) * 4;        // first of 4 global n-tiles
  int colBase = ntBase * 16;

  const uint16_t* arow = A + (size_t)(mt * 16 + lh) * K;
  v8f acc0 = {}, acc1 = {}, acc2 = {}, acc3 = {};

#pragma unroll
  for (int kk = 0; kk < K / 32; ++kk) {
    const int kbase = kk * 32;
    ABFrag a;
#pragma unroll
    for (int v = 0; v < 8; ++v) {             // contiguous k order: k = 16*half + 2v (matches packed B)
      int k = kbase + half * 16 + 2 * v;
      a.u[v] = *(const uint32_t*)(arow + k);
    }
    ABFrag b0, b1, b2, b3;
    const uint4* bq = (const uint4*)(Bp + ((size_t)(kk * 16 + ntBase) * 32 + lane) * 8);
    b0.q[0] = bq[0];   b0.q[1] = bq[1];       // each n-tile block is 32*8 dwords = 64 uint4
    b1.q[0] = bq[64];  b1.q[1] = bq[65];
    b2.q[0] = bq[128]; b2.q[1] = bq[129];
    b3.q[0] = bq[192]; b3.q[1] = bq[193];
    acc0 = __builtin_amdgcn_wmma_f32_16x16x32_bf16(false, a.v, false, b0.v, (short)0, acc0, false, false);
    acc1 = __builtin_amdgcn_wmma_f32_16x16x32_bf16(false, a.v, false, b1.v, (short)0, acc1, false, false);
    acc2 = __builtin_amdgcn_wmma_f32_16x16x32_bf16(false, a.v, false, b2.v, (short)0, acc2, false, false);
    acc3 = __builtin_amdgcn_wmma_f32_16x16x32_bf16(false, a.v, false, b3.v, (short)0, acc3, false, false);
  }

  float* crow = C + (size_t)mt * 16 * NW + colBase + lh;
#pragma unroll
  for (int r = 0; r < 8; ++r) {               // C/D layout: M = r + 8*half, N = lh
    size_t o = (size_t)(r + 8 * half) * NW;
    crow[o +  0] = acc0[r];
    crow[o + 16] = acc1[r];
    crow[o + 32] = acc2[r];
    crow[o + 48] = acc3[r];
  }
}

// ---------------- message passing ----------------
// agg[i] = hw[i] * dinv[i]^2  (self-loop contribution; also serves as init)
__global__ void k_agg_init(const float* __restrict__ hw, const float* __restrict__ dinv,
                           float* __restrict__ agg) {
  int i = blockIdx.x * blockDim.x + threadIdx.x;     // NNODES*64 threads, 4 ch each
  if (i >= NNODES * 64) return;
  int node = i >> 6;
  int c    = (i & 63) * 4;
  float w = dinv[node] * dinv[node];
  const float4 v = *(const float4*)(hw + (size_t)node * CH + c);
  float4 o; o.x = v.x * w; o.y = v.y * w; o.z = v.z * w; o.w = v.w * w;
  *(float4*)(agg + (size_t)node * CH + c) = o;
}

// one wave per edge: agg[dst] += hw[src] * dinv[src]*dinv[dst]   (f32 atomics -> L2)
__global__ void k_edge_scatter(const long long* __restrict__ ei,
                               const float* __restrict__ dinv,
                               const float* __restrict__ hw,
                               float* __restrict__ agg, int E) {
  int wid  = (blockIdx.x * blockDim.x + threadIdx.x) >> 5;
  int lane = threadIdx.x & 31;
  if (wid >= E) return;
  int s = (int)ei[wid];
  int d = (int)ei[E + wid];
  float w = dinv[s] * dinv[d];
  const float4* hs = (const float4*)(hw + (size_t)s * CH);
  float* ad = agg + (size_t)d * CH;
#pragma unroll
  for (int j = 0; j < 2; ++j) {
    int q = lane + j * 32;                    // 64 float4 = 256 channels
    float4 v = hs[q];
    int c = q * 4;
    atomicAdd(ad + c + 0, v.x * w);
    atomicAdd(ad + c + 1, v.y * w);
    atomicAdd(ad + c + 2, v.z * w);
    atomicAdd(ad + c + 3, v.w * w);
  }
}

// ---------------- activations ----------------
// layer-1: relu(agg + b) -> bf16 pairs (feeds next WMMA GEMM)
__global__ void k_bias_relu_bf16(const float* __restrict__ agg, const float* __restrict__ b,
                                 uint32_t* __restrict__ hbf) {
  int i = blockIdx.x * blockDim.x + threadIdx.x;     // NNODES*128 pair-threads
  if (i >= NNODES * (CH / 2)) return;
  int node = i >> 7;
  int c    = (i & 127) * 2;
  float v0 = fmaxf(agg[(size_t)node * CH + c]     + b[c],     0.0f);
  float v1 = fmaxf(agg[(size_t)node * CH + c + 1] + b[c + 1], 0.0f);
  hbf[i] = f2bf(v0) | (f2bf(v1) << 16);
}
// layer-2: relu(agg + b) in place (stays f32 for pooling)
__global__ void k_bias_relu_f32(float* __restrict__ agg, const float* __restrict__ b) {
  int i = blockIdx.x * blockDim.x + threadIdx.x;
  if (i >= NNODES * CH) return;
  agg[i] = fmaxf(agg[i] + b[i & (CH - 1)], 0.0f);
}

// ---------------- pooling + dense head ----------------
__global__ void k_zero_pool(float* __restrict__ sums, float* __restrict__ counts) {
  int i = blockIdx.x * blockDim.x + threadIdx.x;
  if (i < NGRAPH * CH) sums[i] = 0.0f;
  if (i < NGRAPH) counts[i] = 0.0f;
}
__global__ void k_count(const long long* __restrict__ batch, float* __restrict__ counts) {
  int i = blockIdx.x * blockDim.x + threadIdx.x;
  if (i >= NNODES) return;
  atomicAdd(&counts[(int)batch[i]], 1.0f);
}
__global__ void k_pool_scatter(const float* __restrict__ h2, const long long* __restrict__ batch,
                               float* __restrict__ sums) {
  int i = blockIdx.x * blockDim.x + threadIdx.x;     // NNODES*64, 4 ch each
  if (i >= NNODES * 64) return;
  int node = i >> 6;
  int c    = (i & 63) * 4;
  int g    = (int)batch[node];
  float4 v = *(const float4*)(h2 + (size_t)node * CH + c);
  float* s = sums + (size_t)g * CH + c;
  atomicAdd(s + 0, v.x); atomicAdd(s + 1, v.y);
  atomicAdd(s + 2, v.z); atomicAdd(s + 3, v.w);
}
__global__ void k_gmean(const float* __restrict__ sums, const float* __restrict__ counts,
                        float* __restrict__ g) {
  int i = blockIdx.x * blockDim.x + threadIdx.x;
  if (i >= NGRAPH * CH) return;
  g[i] = sums[i] / fmaxf(counts[i >> 8], 1.0f);
}
__global__ void k_dense3(const float* __restrict__ g, const float* __restrict__ W3,
                         const float* __restrict__ b3, float* __restrict__ t3) {
  int i = blockIdx.x * blockDim.x + threadIdx.x;     // 64*128
  if (i >= NGRAPH * 128) return;
  int gi = i >> 7, j = i & 127;
  float acc = b3[j];
  for (int k = 0; k < 256; ++k) acc = fmaf(g[gi * 256 + k], W3[k * 128 + j], acc);
  t3[i] = fmaxf(acc, 0.0f);
}
__global__ void k_dense4(const float* __restrict__ t3, const float* __restrict__ W4,
                         const float* __restrict__ b4, float* __restrict__ out) {
  int i = blockIdx.x * blockDim.x + threadIdx.x;     // 64*10
  if (i >= NGRAPH * 10) return;
  int gi = i / 10, j = i % 10;
  float acc = b4[j];
  for (int k = 0; k < 128; ++k) acc = fmaf(t3[gi * 128 + k], W4[k * 10 + j], acc);
  out[i] = acc;
}

// ---------------- host orchestration ----------------
extern "C" void kernel_launch(void* const* d_in, const int* in_sizes, int n_in,
                              void* d_out, int out_size, void* d_ws, size_t ws_size,
                              hipStream_t stream) {
  const float*     x     = (const float*)d_in[0];
  const long long* ei    = (const long long*)d_in[1];   // int64 [2][E]
  const long long* batch = (const long long*)d_in[2];   // int64 [N]
  const float* W1 = (const float*)d_in[3];  const float* b1 = (const float*)d_in[4];
  const float* W2 = (const float*)d_in[5];  const float* b2 = (const float*)d_in[6];
  const float* W3 = (const float*)d_in[7];  const float* b3 = (const float*)d_in[8];
  const float* W4 = (const float*)d_in[9];  const float* b4 = (const float*)d_in[10];
  float* out = (float*)d_out;
  (void)in_sizes; (void)n_in; (void)out_size; (void)ws_size;

  // workspace layout (~142 MB)
  char* ws = (char*)d_ws; size_t off = 0;
  auto alloc = [&](size_t bytes) -> char* {
    char* p = ws + off; off = (off + bytes + 255) & ~(size_t)255; return p;
  };
  float*    hw   = (float*)   alloc(sizeof(float)    * (size_t)NNODES * CH);   // GEMM out (both layers)
  float*    agg  = (float*)   alloc(sizeof(float)    * (size_t)NNODES * CH);   // scatter target
  uint32_t* hbf  = (uint32_t*)alloc(sizeof(uint16_t) * (size_t)NNODES * CH);   // bf16 h1
  uint32_t* xbf  = (uint32_t*)alloc(sizeof(uint16_t) * (size_t)NNODES * 128);  // bf16 x
  uint32_t* w1p  = (uint32_t*)alloc(sizeof(uint32_t) * (128 / 2) * 256);
  uint32_t* w2p  = (uint32_t*)alloc(sizeof(uint32_t) * (256 / 2) * 256);
  float*    deg  = (float*)   alloc(sizeof(float) * NNODES);
  float*    dinv = (float*)   alloc(sizeof(float) * NNODES);
  float*    sums = (float*)   alloc(sizeof(float) * NGRAPH * CH);
  float*    cnts = (float*)   alloc(sizeof(float) * NGRAPH);
  float*    g    = (float*)   alloc(sizeof(float) * NGRAPH * CH);
  float*    t3   = (float*)   alloc(sizeof(float) * NGRAPH * 128);

  const int B = 256;
  auto blocks = [](long long n, int b) { return (int)((n + b - 1) / b); };

  // precision conversion + fragment-contiguous weight packing
  k_f32_to_bf16<<<blocks((long long)NNODES * 64, B), B, 0, stream>>>(x, xbf, NNODES * 64);
  k_pack_w_frag<<<blocks(64 * 256, B), B, 0, stream>>>(W1, w1p, 128, 256);
  k_pack_w_frag<<<blocks(128 * 256, B), B, 0, stream>>>(W2, w2p, 256, 256);

  // symmetric normalization
  k_deg_init<<<blocks(NNODES, B), B, 0, stream>>>(deg, NNODES);
  k_deg_edges<<<blocks(NEDGES, B), B, 0, stream>>>(ei, deg, NEDGES);
  k_dinv<<<blocks(NNODES, B), B, 0, stream>>>(deg, dinv, NNODES);

  const int gemmWaves  = (NNODES / 16) * (CH / 64);              // 12500
  const int gemmBlocks = blocks((long long)gemmWaves * 32, B);   // 1563
  const int edgeBlocks = blocks((long long)NEDGES * 32, B);      // 75000

  // ---- layer 1 ----
  k_wmma_gemm<128><<<gemmBlocks, B, 0, stream>>>((const uint16_t*)xbf, w1p, hw, NNODES);
  k_agg_init<<<blocks((long long)NNODES * 64, B), B, 0, stream>>>(hw, dinv, agg);
  k_edge_scatter<<<edgeBlocks, B, 0, stream>>>(ei, dinv, hw, agg, NEDGES);
  k_bias_relu_bf16<<<blocks((long long)NNODES * 128, B), B, 0, stream>>>(agg, b1, hbf);

  // ---- layer 2 ----
  k_wmma_gemm<256><<<gemmBlocks, B, 0, stream>>>((const uint16_t*)hbf, w2p, hw, NNODES);
  k_agg_init<<<blocks((long long)NNODES * 64, B), B, 0, stream>>>(hw, dinv, agg);
  k_edge_scatter<<<edgeBlocks, B, 0, stream>>>(ei, dinv, hw, agg, NEDGES);
  k_bias_relu_f32<<<blocks((long long)NNODES * CH, B), B, 0, stream>>>(agg, b2);

  // ---- pooling + dense head ----
  k_zero_pool<<<blocks(NGRAPH * CH, B), B, 0, stream>>>(sums, cnts);
  k_count<<<blocks(NNODES, B), B, 0, stream>>>(batch, cnts);
  k_pool_scatter<<<blocks((long long)NNODES * 64, B), B, 0, stream>>>(agg, batch, sums);
  k_gmean<<<blocks(NGRAPH * CH, B), B, 0, stream>>>(sums, cnts, g);
  k_dense3<<<blocks(NGRAPH * 128, B), B, 0, stream>>>(g, W3, b3, t3);
  k_dense4<<<blocks(NGRAPH * 10, B), B, 0, stream>>>(t3, W4, b4, out);
}
